// MultiHeadedSelfAttention_82068235092504
// MI455X (gfx1250) — matrix operand order repaired and checked by reading, not compiled, not run on hardware
//
#include <hip/hip_runtime.h>
#include <hip/hip_bf16.h>

// MultiHeadedSelfAttention for MI455X (gfx1250, wave32, WMMA).
// B=4, S=2048, HID=1024, H=16, D=64.
// d_out = [h : B*S*HID f32][scores : B*H*S*S f32]
// d_ws  = qp/kp bf16 [B,H,S,D] + vt bf16 [B,H,D,S]  (~50 MB)

#define B_    4
#define S_    2048
#define H_    16
#define D_    64
#define HID_  1024
#define NELEM (B_ * H_ * S_ * D_)   // 8388608 per projection

typedef __bf16 bf16_t;
typedef __attribute__((ext_vector_type(16))) __bf16 v16bf;
typedef __attribute__((ext_vector_type(8)))  __bf16 v8bf;
typedef __attribute__((ext_vector_type(8)))  float  v8f;

union Frag16 { v16bf v; v8bf h[2]; };

__device__ __forceinline__ v8f wmma_bf16(v16bf a, v16bf b, v8f c) {
  // v_wmma_f32_16x16x32_bf16: D = A(16x32) * B(32x16) + C(16x16 f32)
  return __builtin_amdgcn_wmma_f32_16x16x32_bf16(
      /*neg_a=*/false, a, /*neg_b=*/false, b,
      /*c_mod=*/(short)0, c, /*reuse_a=*/false, /*reuse_b=*/false);
}

__device__ __forceinline__ v8bf cvt8(float4 a, float4 b) {
  v8bf r;
  r[0] = (__bf16)a.x; r[1] = (__bf16)a.y; r[2] = (__bf16)a.z; r[3] = (__bf16)a.w;
  r[4] = (__bf16)b.x; r[5] = (__bf16)b.y; r[6] = (__bf16)b.z; r[7] = (__bf16)b.w;
  return r;
}

// ---------------------------------------------------------------------------
// Kernel 1: P = X @ W^T + bias, stored bf16 head-split.
//   grid = (M/128, N/128, 3); block = 256 (8 waves = 4M x 2N),
//   wave tile 32x64 -> 8 WMMAs per K-step from 2 A-frags + 4 B-frags.
// ---------------------------------------------------------------------------
#define ALD 40   // 32 + 8 pad (80B row stride: 16B-aligned, bank-spread)

__global__ __launch_bounds__(256) void proj_kernel(
    const float* __restrict__ q,  const float* __restrict__ k,  const float* __restrict__ v,
    const float* __restrict__ Wq, const float* __restrict__ bq,
    const float* __restrict__ Wk, const float* __restrict__ bk,
    const float* __restrict__ Wv, const float* __restrict__ bv,
    bf16_t* __restrict__ qp, bf16_t* __restrict__ kp, bf16_t* __restrict__ vt) {
  __shared__ __bf16 As[128][ALD];
  __shared__ __bf16 Bs[128][ALD];

  const float* X; const float* W; const float* bias; bf16_t* dst; int transposed = 0;
  if (blockIdx.z == 0)      { X = q; W = Wq; bias = bq; dst = qp; }
  else if (blockIdx.z == 1) { X = k; W = Wk; bias = bk; dst = kp; }
  else                      { X = v; W = Wv; bias = bv; dst = vt; transposed = 1; }

  const int m0 = blockIdx.x * 128;
  const int n0 = blockIdx.y * 128;
  const int t = threadIdx.x;
  const int lane = t & 31, w = t >> 5;
  const int wm = w & 3, wn = w >> 2;      // wave grid 4(M) x 2(N)
  const int l15 = lane & 15, g = lane >> 4;

  v8f acc[2][4] = {};                     // 32(M) x 64(N) per wave

  const int srow = t >> 1, scb = (t & 1) * 16;   // stage: 128 rows x 32 cols
  const float* aptr = X + (size_t)(m0 + srow) * HID_ + scb;
  const float* bptr = W + (size_t)(n0 + srow) * HID_ + scb;

  for (int kt = 0; kt < HID_; kt += 32) {
    const float4* asrc = (const float4*)(aptr + kt);
    float4 a0 = asrc[0], a1 = asrc[1], a2 = asrc[2], a3 = asrc[3];
    *(v8bf*)&As[srow][scb]     = cvt8(a0, a1);
    *(v8bf*)&As[srow][scb + 8] = cvt8(a2, a3);
    const float4* bsrc = (const float4*)(bptr + kt);
    float4 b0 = bsrc[0], b1 = bsrc[1], b2 = bsrc[2], b3 = bsrc[3];
    *(v8bf*)&Bs[srow][scb]     = cvt8(b0, b1);
    *(v8bf*)&Bs[srow][scb + 8] = cvt8(b2, b3);
    if (kt + 32 < HID_) {   // -> global_prefetch_b8, near-cache locality
      __builtin_prefetch(aptr + kt + 32, 0, 3);
      __builtin_prefetch(bptr + kt + 32, 0, 3);
    }
    __syncthreads();

    // A frags: lane row = l15, K chunks {g*8..+7} and {16+g*8..+7}
    Frag16 fa[2], fb[4];
    const int r0 = wm * 32 + l15;
#pragma unroll
    for (int mf = 0; mf < 2; ++mf) {
      fa[mf].h[0] = *(const v8bf*)&As[r0 + mf * 16][g * 8];
      fa[mf].h[1] = *(const v8bf*)&As[r0 + mf * 16][16 + g * 8];
    }
    // B frags: lane col = l15, K chunks {g*8..+7} and {16+g*8..+7} (row of W = col of W^T)
    const int col0 = wn * 64 + l15;
#pragma unroll
    for (int nf = 0; nf < 4; ++nf) {
      fb[nf].h[0] = *(const v8bf*)&Bs[col0 + nf * 16][g * 8];
      fb[nf].h[1] = *(const v8bf*)&Bs[col0 + nf * 16][16 + g * 8];
    }
#pragma unroll
    for (int mf = 0; mf < 2; ++mf)
#pragma unroll
      for (int nf = 0; nf < 4; ++nf)
        acc[mf][nf] = wmma_bf16(fa[mf].v, fb[nf].v, acc[mf][nf]);
    __syncthreads();
  }

  // Layout decision hoisted out of the element loop (uniform branch, clean
  // address streams; transposed V packs 8 bf16 into one 16B store).
#pragma unroll
  for (int mf = 0; mf < 2; ++mf)
#pragma unroll
    for (int nf = 0; nf < 4; ++nf) {
      const v8f cc = acc[mf][nf];
      const int ng = n0 + wn * 64 + nf * 16 + l15;
      const int hh = ng >> 6, d = ng & 63;
      const float bb = bias[ng];
      const int mg0 = m0 + wm * 32 + mf * 16 + g * 8;  // 8 rows, same batch
      const int bi = mg0 >> 11, s0 = mg0 & (S_ - 1);
      if (!transposed) {
        bf16_t* p = dst + (((size_t)bi * H_ + hh) * S_ + s0) * D_ + d;
#pragma unroll
        for (int r = 0; r < 8; ++r) p[(size_t)r * D_] = (bf16_t)(cc[r] + bb);
      } else {
        bf16_t* p = dst + (((size_t)bi * H_ + hh) * D_ + d) * S_ + s0;
        v8bf pk;
#pragma unroll
        for (int r = 0; r < 8; ++r) pk[r] = (bf16_t)(cc[r] + bb);
        *(v8bf*)p = pk;   // s0 multiple of 8 -> 16B aligned
      }
    }
}

// ---------------------------------------------------------------------------
// Kernel 2: per (b, h, 32-query tile): scores = softmax(QK^T/8 + mask),
// h = scores @ V. Full 32x2048 score strip lives in LDS (CDNA5: 320KB/WGP).
// Normalized scores are streamed LDS->global with async stores, overlapped
// with the P@V WMMA loop.
//   grid = (S/32, H, B); block = 256 (8 waves).
// ---------------------------------------------------------------------------
#define SC_LD 2052   // 2048 + 4 pad (row stride 8208B: 16B aligned, bank-spread)

__global__ __launch_bounds__(256) void attn_kernel(
    const bf16_t* __restrict__ qp, const bf16_t* __restrict__ kp,
    const bf16_t* __restrict__ vt, const int* __restrict__ mask,
    float* __restrict__ out_h, float* __restrict__ out_sc) {
  extern __shared__ float smem[];
  float (*sc)[SC_LD] = (float (*)[SC_LD])smem;       // 32 x 2052 f32
  float* red = smem + 32 * SC_LD;                    // 32 x 8 partials

  const int qbase = blockIdx.x * 32;
  const int head  = blockIdx.y;
  const int b     = blockIdx.z;

  const bf16_t* Q  = qp + (((size_t)b * H_ + head) * S_) * D_;
  const bf16_t* Kh = kp + (((size_t)b * H_ + head) * S_) * D_;
  const bf16_t* Vh = vt + (((size_t)b * H_ + head) * D_) * S_;

  const int t = threadIdx.x;
  const int lane = t & 31, w = t >> 5;
  const int l15 = lane & 15, g = lane >> 4;

  // ---- Phase 1: raw scores -> LDS. Wave w owns keys [w*256, w*256+256). ----
  Frag16 qa[2][2];  // [M tile][d-step], kept in registers for the whole strip
#pragma unroll
  for (int mf = 0; mf < 2; ++mf)
#pragma unroll
    for (int ks = 0; ks < 2; ++ks) {
      const int row = qbase + mf * 16 + l15;
      qa[mf][ks].h[0] = *(const v8bf*)&Q[(size_t)row * D_ + ks * 32 + g * 8];
      qa[mf][ks].h[1] = *(const v8bf*)&Q[(size_t)row * D_ + ks * 32 + 16 + g * 8];
    }

  const int keystrip = w * 256;
  for (int kc = 0; kc < 16; ++kc) {
    const int keyl = keystrip + kc * 16 + l15;
    Frag16 kb0, kb1;  // B frag: lane col = key, K(d) = ks*32 + {g*8..+7, 16+g*8..+7}
    kb0.h[0] = *(const v8bf*)&Kh[(size_t)keyl * D_ + g * 8];
    kb0.h[1] = *(const v8bf*)&Kh[(size_t)keyl * D_ + 16 + g * 8];
    kb1.h[0] = *(const v8bf*)&Kh[(size_t)keyl * D_ + 32 + g * 8];
    kb1.h[1] = *(const v8bf*)&Kh[(size_t)keyl * D_ + 48 + g * 8];
    if (kc + 1 < 16)   // prefetch next chunk's key row
      __builtin_prefetch(&Kh[(size_t)(keyl + 16) * D_], 0, 3);
    v8f c0 = {}, c1 = {};
    c0 = wmma_bf16(qa[0][0].v, kb0.v, c0);
    c0 = wmma_bf16(qa[0][1].v, kb1.v, c0);
    c1 = wmma_bf16(qa[1][0].v, kb0.v, c1);
    c1 = wmma_bf16(qa[1][1].v, kb1.v, c1);
    const float madd = -10000.0f * (1.0f - (float)mask[b * S_ + keyl]);
#pragma unroll
    for (int r = 0; r < 8; ++r) {
      sc[g * 8 + r][keyl]      = c0[r] * 0.125f + madd;
      sc[16 + g * 8 + r][keyl] = c1[r] * 0.125f + madd;
    }
  }
  __syncthreads();

  // ---- Phase 2: softmax in LDS (8 threads per row, 256-wide strips) ----
  const int row = t >> 3, seg = t & 7;
  float4* rp4 = (float4*)&sc[row][seg * 256];
  float mx = -3.4e38f;
  for (int i = 0; i < 64; ++i) {
    const float4 x = rp4[i];
    mx = fmaxf(mx, fmaxf(fmaxf(x.x, x.y), fmaxf(x.z, x.w)));
  }
  red[row * 8 + seg] = mx;
  __syncthreads();
  float rmax = red[row * 8];
#pragma unroll
  for (int j = 1; j < 8; ++j) rmax = fmaxf(rmax, red[row * 8 + j]);
  __syncthreads();
  float sum = 0.0f;
  for (int i = 0; i < 64; ++i) {
    float4 x = rp4[i];
    x.x = __expf(x.x - rmax); x.y = __expf(x.y - rmax);
    x.z = __expf(x.z - rmax); x.w = __expf(x.w - rmax);
    rp4[i] = x;               // exp in place (thread-exclusive strip)
    sum += (x.x + x.y) + (x.z + x.w);
  }
  red[row * 8 + seg] = sum;
  __syncthreads();
  float rsum = 0.0f;
#pragma unroll
  for (int j = 0; j < 8; ++j) rsum += red[row * 8 + j];
  const float inv = 1.0f / rsum;
  for (int i = 0; i < 64; ++i) {   // normalize in place
    float4 x = rp4[i];
    x.x *= inv; x.y *= inv; x.z *= inv; x.w *= inv;
    rp4[i] = x;
  }
  __syncthreads();   // all 32 rows normalized and visible

  // ---- Async store: normalized scores LDS -> global (ASYNCcnt-tracked,
  // no VGPR round-trip), overlapped with the P@V WMMA loop below. ----
  float* osc = out_sc + (((size_t)b * H_ + head) * S_ + qbase) * S_;
  {
    const uint32_t lds0 = (uint32_t)(uintptr_t)&sc[0][0];
    for (int i = 0; i < 64; ++i) {
      const int e4 = i * 256 + t;            // float4 index in 32 x 512
      const int r2 = e4 >> 9, c4 = e4 & 511;
      const uint32_t laddr = lds0 + (uint32_t)(r2 * SC_LD + c4 * 4) * 4u;
      float* gp = osc + (size_t)r2 * S_ + c4 * 4;
      asm volatile("global_store_async_from_lds_b128 %0, %1, off"
                   :: "v"(gp), "v"(laddr)
                   : "memory");
    }
  }

  // ---- Phase 3: h = P @ V. Wave grid 2(M) x 4(N); one 16x16 C frag each ----
  const int mf = w & 1, nf = w >> 1;
  const int arow = mf * 16 + l15;     // A frag row (constant per lane)
  const int dcol = nf * 16 + l15;     // B frag col = output d
  const bf16_t* vrow = Vh + (size_t)dcol * S_;  // row d of V^T: contiguous in k
  v8f hc = {};
  for (int ks = 0; ks < 64; ++ks) {
    const int kb = ks * 32;
    const float4 f0 = *(const float4*)&sc[arow][kb + g * 8];
    const float4 f1 = *(const float4*)&sc[arow][kb + g * 8 + 4];
    const float4 f2 = *(const float4*)&sc[arow][kb + 16 + g * 8];
    const float4 f3 = *(const float4*)&sc[arow][kb + 16 + g * 8 + 4];
    Frag16 af;
    af.h[0] = cvt8(f0, f1);   // already normalized
    af.h[1] = cvt8(f2, f3);
    Frag16 bfv;
    bfv.h[0] = *(const v8bf*)&vrow[kb + g * 8];
    bfv.h[1] = *(const v8bf*)&vrow[kb + 16 + g * 8];
    hc = wmma_bf16(af.v, bfv.v, hc);
  }
  float* oh = out_h + ((size_t)b * S_ + qbase) * HID_ + head * D_;
#pragma unroll
  for (int r = 0; r < 8; ++r) {
    const int rr = mf * 16 + g * 8 + r;
    oh[(size_t)rr * HID_ + nf * 16 + l15] = hc[r];
  }

  asm volatile("s_wait_asynccnt 0x0" ::: "memory");  // drain async stores
}

// ---------------------------------------------------------------------------
extern "C" void kernel_launch(void* const* d_in, const int* in_sizes, int n_in,
                              void* d_out, int out_size, void* d_ws, size_t ws_size,
                              hipStream_t stream) {
  (void)in_sizes; (void)n_in; (void)out_size; (void)ws_size;
  const float* q  = (const float*)d_in[0];
  const float* k  = (const float*)d_in[1];
  const float* v  = (const float*)d_in[2];
  const float* Wq = (const float*)d_in[3];
  const float* bq = (const float*)d_in[4];
  const float* Wk = (const float*)d_in[5];
  const float* bk = (const float*)d_in[6];
  const float* Wv = (const float*)d_in[7];
  const float* bv = (const float*)d_in[8];
  const int* mask = (const int*)d_in[9];

  bf16_t* qp = (bf16_t*)d_ws;
  bf16_t* kp = qp + NELEM;
  bf16_t* vt = kp + NELEM;

  float* out_h  = (float*)d_out;
  float* out_sc = out_h + (size_t)B_ * S_ * HID_;

  proj_kernel<<<dim3(64, 8, 3), 256, 0, stream>>>(
      q, k, v, Wq, bq, Wk, bk, Wv, bv, qp, kp, vt);

  const size_t shm = (size_t)(32 * SC_LD + 32 * 8) * sizeof(float);  // ~257 KB
  hipFuncSetAttribute((const void*)attn_kernel,
                      hipFuncAttributeMaxDynamicSharedMemorySize, (int)shm);
  attn_kernel<<<dim3(S_ / 32, H_, B_), 256, shm, stream>>>(
      qp, kp, vt, mask, out_h, out_sc);
}
